// ArcFaceInnerProduct_47897475285017
// MI455X (gfx1250) — matrix operand order repaired and checked by reading, not compiled, not run on hardware
//
#include <hip/hip_runtime.h>
#include <hip/hip_bf16.h>

// ---------------------------------------------------------------------------
// ArcFace inner product for MI455X (gfx1250, wave32, WMMA + async-to-LDS).
//   cos      = norm(feat) @ norm(weights)^T                [4096 x 20000]
//   marginal = SCALE*cos everywhere, except at (r,label[r]) (tiny fixup pass)
// GEMM: bf16x3 split (hi/lo) on v_wmma_f32_16x16x32_bf16, 128x128 block tile
// staged through LDS with global_load_async_to_lds_b128 (ASYNCcnt pipelined,
// double buffered). Output bandwidth (655 MB @ 23.3 TB/s ~ 28 us) is the floor.
// ---------------------------------------------------------------------------

typedef __attribute__((ext_vector_type(16))) __bf16 v16bf;
typedef __attribute__((ext_vector_type(8)))  __bf16 v8bf;
typedef __attribute__((ext_vector_type(8)))  float  v8f;

#define SCALE_F   30.0f
#define COS_M     0.87758256189037276f
#define SIN_M     0.47942553860420301f
#define THRESH_F (-0.87758256189037276f)
#define EXT_F    (-0.23971276930210151f)   // -0.5*sin(0.5)

__device__ __forceinline__ unsigned short f2bf(float f) {
  unsigned u = __float_as_uint(f);
  u += 0x7FFFu + ((u >> 16) & 1u);
  return (unsigned short)(u >> 16);
}

// --------------------------------------------------------------------------
// Row L2-normalize + split into bf16 hi/lo pair. One wave per row (K=512).
// --------------------------------------------------------------------------
__global__ __launch_bounds__(256, 1)
void normsplit_kernel(const float* __restrict__ X,
                      unsigned short* __restrict__ Hi,
                      unsigned short* __restrict__ Lo,
                      int rows)
{
  const int lane = threadIdx.x & 31;
  const int row  = blockIdx.x * 8 + (threadIdx.x >> 5);
  if (row >= rows) return;

  const float* x = X + (size_t)row * 512;
  float v[16];
  float ss = 0.0f;
#pragma unroll
  for (int i = 0; i < 16; ++i) {
    v[i] = x[i * 32 + lane];
    ss += v[i] * v[i];
  }
#pragma unroll
  for (int off = 16; off > 0; off >>= 1)
    ss += __shfl_xor(ss, off, 32);
  const float rn = rsqrtf(ss);

#pragma unroll
  for (int i = 0; i < 16; ++i) {
    const float y = v[i] * rn;
    const unsigned short h = f2bf(y);
    const float hf = __uint_as_float((unsigned)h << 16);
    const unsigned short l = f2bf(y - hf);
    Hi[(size_t)row * 512 + i * 32 + lane] = h;
    Lo[(size_t)row * 512 + i * 32 + lane] = l;
  }
}

// --------------------------------------------------------------------------
// WMMA GEMM: 256 threads = 8 waves (2x4), block tile 128x128, wave tile 64x32.
// Stage tiles (128 rows x 32 K of Ahi/Alo/Bhi/Blo) live in LDS, filled by
// global_load_async_to_lds_b128 (16B/lane chunks), double buffered on ASYNCcnt.
// LDS rows padded to 80B: 20-bank stride => conflict-free ds_load_b128.
// --------------------------------------------------------------------------
__global__ __launch_bounds__(256, 1)
void arcface_gemm_kernel(const __bf16* __restrict__ Ah, const __bf16* __restrict__ Al,
                         const __bf16* __restrict__ Wh, const __bf16* __restrict__ Wl,
                         float* __restrict__ cosOut, float* __restrict__ margOut,
                         int N)
{
  constexpr int K  = 512;
  constexpr int RS = 40;                       // LDS row stride in bf16 (80 B)
  __shared__ __align__(16) __bf16 smem[2][4][128 * RS];   // [stage][Ah,Al,Bh,Bl]

  const int tid  = threadIdx.x;
  const int lane = tid & 31;
  const int wave = tid >> 5;
  const int wm   = wave >> 2;                  // 0..1
  const int wn   = wave & 3;                   // 0..3
  const int rowTile = blockIdx.y * 128;
  const int colTile = blockIdx.x * 128;
  const int lh  = lane & 15;
  const int sel = lane >> 4;                   // 0 or 1 (half-wave select)

  // Issue one stage: 4 tiles x 128 rows x 64B, as 16B chunks. 512 chunks per
  // tile, 2 per thread per tile -> 8 async instructions per wave per stage.
  auto issue_stage = [&](int b, int kk) {
#pragma unroll
    for (int j = 0; j < 2; ++j) {
      const int c    = tid + j * 256;          // 0..511
      const int row  = c >> 2;                 // 0..127
      const int part = c & 3;                  // 16B chunk within 64B row slab
      const unsigned eoff0 = (unsigned)(uintptr_t)&smem[b][0][row * RS + part * 8];
      const unsigned eoff1 = (unsigned)(uintptr_t)&smem[b][1][row * RS + part * 8];
      const unsigned eoff2 = (unsigned)(uintptr_t)&smem[b][2][row * RS + part * 8];
      const unsigned eoff3 = (unsigned)(uintptr_t)&smem[b][3][row * RS + part * 8];
      const size_t ga = (size_t)(rowTile + row) * K + kk + part * 8;
      const int wr = colTile + row;
      const int wrc = wr < N ? wr : N - 1;     // clamp on N remainder block
      const size_t gb = (size_t)wrc * K + kk + part * 8;
      asm volatile("global_load_async_to_lds_b128 %0, %1, off"
                   :: "v"(eoff0), "v"((unsigned long long)(uintptr_t)(Ah + ga))
                   : "memory");
      asm volatile("global_load_async_to_lds_b128 %0, %1, off"
                   :: "v"(eoff1), "v"((unsigned long long)(uintptr_t)(Al + ga))
                   : "memory");
      asm volatile("global_load_async_to_lds_b128 %0, %1, off"
                   :: "v"(eoff2), "v"((unsigned long long)(uintptr_t)(Wh + gb))
                   : "memory");
      asm volatile("global_load_async_to_lds_b128 %0, %1, off"
                   :: "v"(eoff3), "v"((unsigned long long)(uintptr_t)(Wl + gb))
                   : "memory");
    }
  };

  v8f acc[4][2] = {};                          // 8 f32 accumulator tiles / wave

  issue_stage(0, 0);

  for (int k = 0; k < K; k += 32) {
    const int buf = (k >> 5) & 1;
    if (k + 32 < K) {
      issue_stage(buf ^ 1, k + 32);
      // 16 outstanding; oldest 8 (= this stage) retire in order.
      asm volatile("s_wait_asynccnt 0x8" ::: "memory");
    } else {
      asm volatile("s_wait_asynccnt 0x0" ::: "memory");
    }
    __syncthreads();                           // stage data visible to all waves

    v16bf ahf[4], alf[4], bhf[2], blf[2];
#pragma unroll
    for (int mi = 0; mi < 4; ++mi) {
      const __bf16* p = &smem[buf][0][(wm * 64 + mi * 16 + lh) * RS + sel * 8];
      v8bf h0 = *(const v8bf*)p;
      v8bf h1 = *(const v8bf*)(p + 16);
      ahf[mi] = __builtin_shufflevector(h0, h1, 0,1,2,3,4,5,6,7,8,9,10,11,12,13,14,15);
      const __bf16* q = &smem[buf][1][(wm * 64 + mi * 16 + lh) * RS + sel * 8];
      v8bf l0 = *(const v8bf*)q;
      v8bf l1 = *(const v8bf*)(q + 16);
      alf[mi] = __builtin_shufflevector(l0, l1, 0,1,2,3,4,5,6,7,8,9,10,11,12,13,14,15);
    }
#pragma unroll
    for (int ni = 0; ni < 2; ++ni) {
      const __bf16* p = &smem[buf][2][(wn * 32 + ni * 16 + lh) * RS + sel * 8];
      v8bf h0 = *(const v8bf*)p;
      v8bf h1 = *(const v8bf*)(p + 16);
      bhf[ni] = __builtin_shufflevector(h0, h1, 0,1,2,3,4,5,6,7,8,9,10,11,12,13,14,15);
      const __bf16* q = &smem[buf][3][(wn * 32 + ni * 16 + lh) * RS + sel * 8];
      v8bf l0 = *(const v8bf*)q;
      v8bf l1 = *(const v8bf*)(q + 16);
      blf[ni] = __builtin_shufflevector(l0, l1, 0,1,2,3,4,5,6,7,8,9,10,11,12,13,14,15);
    }

    // bf16x3 split product: Ahi*Bhi + Ahi*Blo + Alo*Bhi  (~fp32 accurate)
#pragma unroll
    for (int mi = 0; mi < 4; ++mi) {
#pragma unroll
      for (int ni = 0; ni < 2; ++ni) {
        acc[mi][ni] = __builtin_amdgcn_wmma_f32_16x16x32_bf16(
            false, ahf[mi], false, bhf[ni], (short)0, acc[mi][ni], false, false);
        acc[mi][ni] = __builtin_amdgcn_wmma_f32_16x16x32_bf16(
            false, ahf[mi], false, blf[ni], (short)0, acc[mi][ni], false, false);
        acc[mi][ni] = __builtin_amdgcn_wmma_f32_16x16x32_bf16(
            false, alf[mi], false, bhf[ni], (short)0, acc[mi][ni], false, false);
      }
    }
    __syncthreads();   // protect this buffer before next iteration's async issue
  }

  // Epilogue: C/D layout -> lane (0-15|16-31) = N col, VGPR r = row (M = r+8*sel)
#pragma unroll
  for (int mi = 0; mi < 4; ++mi) {
#pragma unroll
    for (int ni = 0; ni < 2; ++ni) {
      const int col = colTile + wn * 32 + ni * 16 + lh;
      if (col < N) {
#pragma unroll
        for (int r = 0; r < 8; ++r) {
          const int row = rowTile + wm * 64 + mi * 16 + sel * 8 + r;
          const float c = acc[mi][ni][r];
          const size_t idx = (size_t)row * N + col;
          cosOut[idx]  = c;
          margOut[idx] = SCALE_F * c;
        }
      }
    }
  }
}

// --------------------------------------------------------------------------
// Per-row fixup of the single label column with the ArcFace margin.
// --------------------------------------------------------------------------
__global__ void arcface_fixup_kernel(const int* __restrict__ label,
                                     const float* __restrict__ cosOut,
                                     float* __restrict__ margOut,
                                     int B, int N)
{
  const int r = blockIdx.x * blockDim.x + threadIdx.x;
  if (r >= B) return;
  const int lab = label[r];
  const size_t idx = (size_t)r * N + lab;
  const float c = cosOut[idx];
  float m;
  if (c > THRESH_F) {
    const float s = sqrtf(fmaxf(0.0f, 1.0f - c * c));
    m = SCALE_F * (c * COS_M - s * SIN_M);   // cos(arccos(c)+M)
  } else {
    m = SCALE_F * (c + EXT_F);
  }
  margOut[idx] = m;
}

// --------------------------------------------------------------------------
extern "C" void kernel_launch(void* const* d_in, const int* in_sizes, int n_in,
                              void* d_out, int out_size, void* d_ws, size_t ws_size,
                              hipStream_t stream) {
  const float* feat    = (const float*)d_in[0];
  const int*   label   = (const int*)d_in[1];   // jnp.int64 w/o x64 -> int32
  const float* weights = (const float*)d_in[2];

  const int B = in_sizes[1];            // 4096
  const int D = in_sizes[0] / B;        // 512
  const int C = in_sizes[2] / D;        // 20000

  float* cosOut  = (float*)d_out;
  float* margOut = cosOut + (size_t)B * C;

  // Scratch layout: Ah | Al | Wh | Wl  (bf16 as ushort), ~49.4 MB total
  unsigned short* Ah = (unsigned short*)d_ws;
  unsigned short* Al = Ah + (size_t)B * D;
  unsigned short* Wh = Al + (size_t)B * D;
  unsigned short* Wl = Wh + (size_t)C * D;

  normsplit_kernel<<<(B + 7) / 8, 256, 0, stream>>>(feat, Ah, Al, B);
  normsplit_kernel<<<(C + 7) / 8, 256, 0, stream>>>(weights, Wh, Wl, C);

  dim3 grid((C + 127) / 128, B / 128);
  arcface_gemm_kernel<<<grid, 256, 0, stream>>>(
      (const __bf16*)Ah, (const __bf16*)Al,
      (const __bf16*)Wh, (const __bf16*)Wl,
      cosOut, margOut, C);

  arcface_fixup_kernel<<<(B + 255) / 256, 256, 0, stream>>>(label, cosOut, margOut, B, C);
}
